// HeteroGNN_12970801234250
// MI455X (gfx1250) — matrix operand order repaired and checked by reading, not compiled, not run on hardware
//
#include <hip/hip_runtime.h>
#include <hip/hip_bf16.h>
#include <cstdint>

typedef __attribute__((ext_vector_type(16))) _Float16 v16h;
typedef __attribute__((ext_vector_type(8)))  _Float16 v8h;
typedef __attribute__((ext_vector_type(8)))  float    v8f;

#define HEADS 4
#define CH    64
#define DIM   256          // K for every GEMM in this model
#define NEG_SLOPE_F 0.2f
#define EPS_F 1e-16f
#define FRAGH (8 * 32 * 16)   // halves of packed B per 16-col tile (all K)

// ---------------------------------------------------------------------------
// f32 -> f16 elementwise convert (activations, once per layer per node type)
// ---------------------------------------------------------------------------
__global__ void f32_to_f16(const float* __restrict__ in,
                           _Float16* __restrict__ out, long n)
{
  long i = (long)blockIdx.x * blockDim.x + threadIdx.x;
  if (i < n) out[i] = (_Float16)in[i];
}

// ---------------------------------------------------------------------------
// Pack B[K=256, N] (f32 row-major) into per-wave WMMA B-fragments (f16):
//   frag id = (tileN*8 + kstep)*32 + lane, 16 halves each (32B contiguous).
//   lane holds column n = tileN*16 + (lane&15), K = kstep*32 + (lane>>4)*8 + j (+8 for j>=8)
// ---------------------------------------------------------------------------
__global__ void pack_b_f16(const float* __restrict__ B,
                           _Float16* __restrict__ out, int N)
{
  const int tid = blockIdx.x * blockDim.x + threadIdx.x;
  const int total = (N >> 4) * 8 * 32;
  if (tid >= total) return;
  const int lane  = tid & 31;
  const int kstep = (tid >> 5) & 7;
  const int tileN = tid >> 8;
  const int n     = tileN * 16 + (lane & 15);
  const int kbase = kstep * 32 + (lane >> 4) * 8;
  _Float16* dst = out + (size_t)tid * 16;
#pragma unroll
  for (int j = 0; j < 16; ++j) dst[j] = (_Float16)B[(size_t)(kbase + j + (j >= 8 ? 8 : 0)) * N + n];
}

// load one K-step's fragments (A 16x32 frag + 4 B frags for 4 col-tiles)
__device__ __forceinline__ void load_frags(
    const _Float16* __restrict__ arow, const _Float16* __restrict__ bb,
    int ks, int kloA, v16h& a, v16h& b0, v16h& b1, v16h& b2, v16h& b3)
{
  const int k0 = ks * 32;
  const v8h alo = *(const v8h*)(arow + k0 + kloA);
  const v8h ahi = *(const v8h*)(arow + k0 + 16 + kloA);
#pragma unroll
  for (int j = 0; j < 8; ++j) { a[j] = alo[j]; a[8 + j] = ahi[j]; }
  const _Float16* p = bb + (size_t)ks * (32 * 16);
  b0 = *(const v16h*)(p);
  b1 = *(const v16h*)(p + FRAGH);
  b2 = *(const v16h*)(p + 2 * FRAGH);
  b3 = *(const v16h*)(p + 3 * FRAGH);
}

// ---------------------------------------------------------------------------
// C[M,N] = maybe_relu(A[M,256] @ B + bias), A f16 row-major, B pre-packed f16.
// One wave computes 16 rows x 64 cols (4 accumulators, A-frag reused 4x).
// K loop fully unrolled (8 steps) and software-pipelined in registers so the
// next step's global_load_b128s are in flight during the current WMMAs.
// ---------------------------------------------------------------------------
__global__ void __launch_bounds__(256) gemm_f16_wmma(
    const _Float16* __restrict__ A, const _Float16* __restrict__ Bp,
    const float* __restrict__ bias, float* __restrict__ C,
    int M, int N, int relu)
{
  const int wave    = blockIdx.x * (blockDim.x >> 5) + (threadIdx.x >> 5);
  const int lane    = threadIdx.x & 31;
  const int groupsN = N >> 6;                 // 64-column groups
  const int tileM   = wave / groupsN;
  const int g       = wave - tileM * groupsN;
  if (tileM * 16 >= M) return;

  const int halfSel = lane >> 4;
  const int l15     = lane & 15;
  const int m       = tileM * 16 + l15;
  const int kloA    = halfSel * 8;            // A: K = klo..klo+7 and 16+klo..
  const int tileN0  = g * 4;

  const _Float16* arow = A + (size_t)m * DIM;
  const _Float16* bb   = Bp + ((size_t)(tileN0 * 8) * 32 + lane) * 16;

  v8f acc0 = {}, acc1 = {}, acc2 = {}, acc3 = {};
  v16h a, b0, b1, b2, b3;
  load_frags(arow, bb, 0, kloA, a, b0, b1, b2, b3);
#pragma unroll
  for (int ks = 0; ks < 8; ++ks) {
    v16h an, b0n, b1n, b2n, b3n;
    if (ks < 7) load_frags(arow, bb, ks + 1, kloA, an, b0n, b1n, b2n, b3n);
    acc0 = __builtin_amdgcn_wmma_f32_16x16x32_f16(false, a, false, b0, (short)0, acc0, false, false);
    acc1 = __builtin_amdgcn_wmma_f32_16x16x32_f16(false, a, false, b1, (short)0, acc1, false, false);
    acc2 = __builtin_amdgcn_wmma_f32_16x16x32_f16(false, a, false, b2, (short)0, acc2, false, false);
    acc3 = __builtin_amdgcn_wmma_f32_16x16x32_f16(false, a, false, b3, (short)0, acc3, false, false);
    a = an; b0 = b0n; b1 = b1n; b2 = b2n; b3 = b3n;
  }

  v8f accs[4] = {acc0, acc1, acc2, acc3};
#pragma unroll
  for (int t = 0; t < 4; ++t) {
    const int n = (tileN0 + t) * 16 + l15;
    const float bv = bias[n];                 // always valid (zero buf if none)
#pragma unroll
    for (int r = 0; r < 8; ++r) {
      const int row = tileM * 16 + r + halfSel * 8;  // C layout: v_r -> M=r/r+8
      float v = accs[t][r] + bv;
      v = relu ? fmaxf(v, 0.0f) : v;
      C[(size_t)row * N + n] = v;
    }
  }
}

// e[n,h] = sum_c h[n, h*64+c] * att[h*64+c]
__global__ void att_dot_kernel(const float* __restrict__ hmat,
                               const float* __restrict__ att,
                               float* __restrict__ e, int N)
{
  const int tid = blockIdx.x * blockDim.x + threadIdx.x;
  if (tid >= N * HEADS) return;
  const int n = tid >> 2, hh = tid & 3;
  const float* hp = hmat + (size_t)n * DIM + hh * CH;
  const float* ap = att + hh * CH;
  float s = 0.f;
#pragma unroll 8
  for (int c = 0; c < CH; ++c) s += hp[c] * ap[c];
  e[tid] = s;
}

__global__ void zero_f32(float* __restrict__ p, long n)
{
  long i = (long)blockIdx.x * blockDim.x + threadIdx.x;
  if (i < n) p[i] = 0.f;
}

// pass 1: un-normalized softmax weights + per-(dst,head) denominator
__global__ void edge_softmax_num(const float* __restrict__ es,
                                 const float* __restrict__ ed,
                                 const int* __restrict__ src,
                                 const int* __restrict__ dst,
                                 float* __restrict__ wbuf,
                                 float* __restrict__ denom, int E)
{
  const int tid = blockIdx.x * blockDim.x + threadIdx.x;
  if (tid >= E * HEADS) return;
  const int e = tid >> 2, h = tid & 3;
  const int s = src[e], d = dst[e];
  float a = es[s * HEADS + h] + ed[d * HEADS + h];
  a = a > 0.f ? a : NEG_SLOPE_F * a;       // leaky_relu
  const float ex = expf(a);                // logits ~|0.5| -> no max pass needed
  wbuf[tid] = ex;
  atomicAdd(&denom[d * HEADS + h], ex);
}

// pass 2: agg[dst,:] += hs[src,:] * w[e, head(c)]  (one block per edge)
__global__ void __launch_bounds__(DIM) edge_aggregate(
    const float* __restrict__ hsrc,
    const int* __restrict__ src, const int* __restrict__ dst,
    const float* __restrict__ wbuf, float* __restrict__ agg)
{
  const int e = blockIdx.x;            // uniform -> scalar loads for indices
  const int c = threadIdx.x;           // 0..255
  const int s = src[e], d = dst[e];
  const float w = wbuf[e * HEADS + (c >> 6)];
  atomicAdd(&agg[(size_t)d * DIM + c], hsrc[(size_t)s * DIM + c] * w);
}

// out = relu(agg / (denom + eps) + bias), in place (one block per node)
__global__ void __launch_bounds__(DIM) finalize_norm(
    float* __restrict__ agg, const float* __restrict__ denom,
    const float* __restrict__ bias)
{
  const int n = blockIdx.x, c = threadIdx.x;
  const size_t idx = (size_t)n * DIM + c;
  float v = agg[idx] / (denom[n * HEADS + (c >> 6)] + EPS_F) + bias[c];
  agg[idx] = v > 0.f ? v : 0.f;
}

// ---------------------------------------------------------------------------
static inline int ceil_div(long a, int b) { return (int)((a + b - 1) / b); }

static void launch_gemm(const _Float16* A, const _Float16* Bp, const float* bias,
                        float* C, int M, int N, int relu, hipStream_t stream)
{
  const long waves = (long)((M + 15) / 16) * (N >> 6);
  gemm_f16_wmma<<<ceil_div(waves, 8), 256, 0, stream>>>(A, Bp, bias, C, M, N, relu);
}

static void launch_pack(const float* B, _Float16* Bp, int N, hipStream_t stream)
{
  const int total = (N >> 4) * 8 * 32;
  pack_b_f16<<<ceil_div(total, 256), 256, 0, stream>>>(B, Bp, N);
}

// one bipartite GATConv relation: (xs_h, xd_h f16) -> new dst features in `agg`
static void run_gat_relation(const _Float16* xs_h, const _Float16* xd_h,
                             const float* Ws, const float* Wd,
                             const float* as, const float* ad,
                             const float* bias, const float* zeroBias,
                             const int* src, const int* dst, int E,
                             int Ns, int Nd,
                             _Float16* packA, _Float16* packB,
                             float* hs, float* hd, float* es, float* ed,
                             float* denom, float* wbuf, float* agg,
                             hipStream_t stream)
{
  launch_pack(Ws, packA, DIM, stream);
  launch_gemm(xs_h, packA, zeroBias, hs, Ns, DIM, 0, stream);
  launch_pack(Wd, packB, DIM, stream);
  launch_gemm(xd_h, packB, zeroBias, hd, Nd, DIM, 0, stream);
  att_dot_kernel<<<ceil_div((long)Ns * HEADS, 256), 256, 0, stream>>>(hs, as, es, Ns);
  att_dot_kernel<<<ceil_div((long)Nd * HEADS, 256), 256, 0, stream>>>(hd, ad, ed, Nd);
  zero_f32<<<ceil_div((long)Nd * HEADS, 256), 256, 0, stream>>>(denom, (long)Nd * HEADS);
  zero_f32<<<ceil_div((long)Nd * DIM, 256), 256, 0, stream>>>(agg, (long)Nd * DIM);
  edge_softmax_num<<<ceil_div((long)E * HEADS, 256), 256, 0, stream>>>(
      es, ed, src, dst, wbuf, denom, E);
  edge_aggregate<<<E, DIM, 0, stream>>>(hs, src, dst, wbuf, agg);
  finalize_norm<<<Nd, DIM, 0, stream>>>(agg, denom, bias);
}

extern "C" void kernel_launch(void* const* d_in, const int* in_sizes, int n_in,
                              void* d_out, int out_size, void* d_ws, size_t ws_size,
                              hipStream_t stream)
{
  const float* x_u     = (const float*)d_in[0];
  const float* x_v     = (const float*)d_in[1];
  const float* W_src   = (const float*)d_in[2];   // [L,2,256,256]
  const float* W_dst   = (const float*)d_in[3];   // [L,2,256,256]
  const float* att_src = (const float*)d_in[4];   // [L,2,4,64]
  const float* att_dst = (const float*)d_in[5];   // [L,2,4,64]
  const float* bias    = (const float*)d_in[6];   // [L,2,256]
  const float* lin_w   = (const float*)d_in[7];   // [256,OUT]
  const float* lin_b   = (const float*)d_in[8];   // [OUT]
  const int*   ei_uv   = (const int*)d_in[9];     // [2,E]: row0=src(u), row1=dst(v)
  const int*   ei_vu   = (const int*)d_in[10];    // [2,E]: row0=src(v), row1=dst(u)

  const int NU   = in_sizes[0] / DIM;
  const int NV   = in_sizes[1] / DIM;
  const int E    = in_sizes[9] / 2;
  const int OUTC = in_sizes[8];
  const int NMAX = NU > NV ? NU : NV;

  // workspace carve-up: f32 region, then f16 region
  float* ws = (float*)d_ws;
  const size_t nodeU = (size_t)NU * DIM, nodeV = (size_t)NV * DIM;
  const size_t nodeM = (size_t)NMAX * DIM;
  float* bufU1 = ws;  ws += nodeU;          // layer-0 output (u)
  float* bufV1 = ws;  ws += nodeV;          // layer-0 output (v)
  float* bufU2 = ws;  ws += nodeU;          // layer-1 output (u)
  float* bufV2 = ws;  ws += nodeV;          // layer-1 output (v)
  float* hs    = ws;  ws += nodeM;          // per-relation src projection
  float* hd    = ws;  ws += nodeM;          // per-relation dst projection
  float* es    = ws;  ws += (size_t)NMAX * HEADS;
  float* ed    = ws;  ws += (size_t)NMAX * HEADS;
  float* denom = ws;  ws += (size_t)NMAX * HEADS;
  float* wbuf  = ws;  ws += (size_t)E * HEADS;
  float* zbias = ws;  ws += DIM;            // always-zero bias vector
  _Float16* hws = (_Float16*)ws;
  _Float16* xhU   = hws;  hws += nodeU;     // f16 activations (u)
  _Float16* xhV   = hws;  hws += nodeV;     // f16 activations (v)
  _Float16* packA = hws;  hws += (size_t)DIM * DIM;  // packed weight frags
  _Float16* packB = hws;  hws += (size_t)DIM * DIM;
  (void)ws_size; (void)n_in; (void)out_size;

  const size_t WSTRIDE = (size_t)DIM * DIM;  // per (layer,relation) weight
  const size_t ASTRIDE = HEADS * CH;         // 256
  const size_t BSTRIDE = DIM;                // 256

  zero_f32<<<1, DIM, 0, stream>>>(zbias, DIM);

  // ---- layer 0 ----
  f32_to_f16<<<ceil_div((long)nodeU, 256), 256, 0, stream>>>(x_u, xhU, (long)nodeU);
  f32_to_f16<<<ceil_div((long)nodeV, 256), 256, 0, stream>>>(x_v, xhV, (long)nodeV);
  // relation u->v : new v features
  run_gat_relation(xhU, xhV,
                   W_src + 0 * WSTRIDE, W_dst + 0 * WSTRIDE,
                   att_src + 0 * ASTRIDE, att_dst + 0 * ASTRIDE,
                   bias + 0 * BSTRIDE, zbias,
                   ei_uv, ei_uv + E, E, NU, NV,
                   packA, packB, hs, hd, es, ed, denom, wbuf, bufV1, stream);
  // relation v->u : new u features
  run_gat_relation(xhV, xhU,
                   W_src + 1 * WSTRIDE, W_dst + 1 * WSTRIDE,
                   att_src + 1 * ASTRIDE, att_dst + 1 * ASTRIDE,
                   bias + 1 * BSTRIDE, zbias,
                   ei_vu, ei_vu + E, E, NV, NU,
                   packA, packB, hs, hd, es, ed, denom, wbuf, bufU1, stream);

  // ---- layer 1 ----
  f32_to_f16<<<ceil_div((long)nodeU, 256), 256, 0, stream>>>(bufU1, xhU, (long)nodeU);
  f32_to_f16<<<ceil_div((long)nodeV, 256), 256, 0, stream>>>(bufV1, xhV, (long)nodeV);
  run_gat_relation(xhU, xhV,
                   W_src + 2 * WSTRIDE, W_dst + 2 * WSTRIDE,
                   att_src + 2 * ASTRIDE, att_dst + 2 * ASTRIDE,
                   bias + 2 * BSTRIDE, zbias,
                   ei_uv, ei_uv + E, E, NU, NV,
                   packA, packB, hs, hd, es, ed, denom, wbuf, bufV2, stream);
  run_gat_relation(xhV, xhU,
                   W_src + 3 * WSTRIDE, W_dst + 3 * WSTRIDE,
                   att_src + 3 * ASTRIDE, att_dst + 3 * ASTRIDE,
                   bias + 3 * BSTRIDE, zbias,
                   ei_vu, ei_vu + E, E, NV, NU,
                   packA, packB, hs, hd, es, ed, denom, wbuf, bufU2, stream);

  // ---- final linear: out = x_u @ lin_w + lin_b ----
  f32_to_f16<<<ceil_div((long)nodeU, 256), 256, 0, stream>>>(bufU2, xhU, (long)nodeU);
  launch_pack(lin_w, packA, OUTC, stream);
  launch_gemm(xhU, packA, lin_b, (float*)d_out, NU, OUTC, 0, stream);
}
